// YatCausalAttention_64793876627785
// MI455X (gfx1250) — matrix operand order repaired and checked by reading, not compiled, not run on hardware
//
#include <hip/hip_runtime.h>

// ---------------------------------------------------------------------------
// Yat causal attention for MI455X (gfx1250, wave32, WMMA bf16).
// Precision: bf16 operands, f32 accumulation (v_wmma_f32_16x16x32_bf16).
// Flash-attention single pass; scores never hit HBM.
// CDNA5 paths: v_wmma bf16, global_load_async_to_lds_b128 staging (ASYNCcnt),
// split wait counters, dual-issue VALU under WMMA.
// Workspace use: 4 x 8MB bf16 buffers (Q,K,V,attn_out) + 512KB norms ~ 34MB.
// ---------------------------------------------------------------------------

#define EMBED  1024
#define NHEADS 16
#define HDIM   64
#define BATCH  2
#define SEQ    2048
#define EPS    1e-6f

typedef __attribute__((ext_vector_type(16))) __bf16 v16bf;
typedef __attribute__((ext_vector_type(8)))  float  v8f;

struct U8x32 { uint4 a, b; };

static __device__ __forceinline__ v16bf frag_from(uint4 a, uint4 b) {
  U8x32 t; t.a = a; t.b = b;
  return __builtin_bit_cast(v16bf, t);
}

// fp32 -> bf16 round-to-nearest-even
static __device__ __forceinline__ unsigned short f2bf(float f) {
  unsigned int u = __builtin_bit_cast(unsigned int, f);
  u += 0x7FFFu + ((u >> 16) & 1u);
  return (unsigned short)(u >> 16);
}
static __device__ __forceinline__ float bf2f(unsigned short h) {
  unsigned int u = ((unsigned int)h) << 16;
  return __builtin_bit_cast(float, u);
}

// Async straight-copy global(16B) -> LDS(16B); tracked on ASYNCcnt.
static __device__ __forceinline__ void async_copy_b128(
    const unsigned short* gsrc, unsigned short* lds_dst) {
  unsigned lds_off = (unsigned)(unsigned long long)lds_dst;
  unsigned long long gaddr = (unsigned long long)gsrc;
  asm volatile("global_load_async_to_lds_b128 %0, %1, off"
               :: "v"(lds_off), "v"(gaddr) : "memory");
}
static __device__ __forceinline__ void wait_async0() {
  asm volatile("s_wait_asynccnt 0" ::: "memory");
}

// ---------------------------------------------------------------------------
// Tiled GEMM: C[M,N] = A[M,K] * B[K,N] + bias.
// MODE 0: A fp32 (x), output scattered as bf16 into Q/K/V [b,h,l,d] buffers.
// MODE 1: A bf16 (attn out), output fp32 to d_out.
// Block: 128 threads (4 waves), 64x64 tile, wave = 32x32 (2x2 WMMA frags).
// ---------------------------------------------------------------------------
#define AS 40   // padded LDS row stride (shorts), keeps 16B alignment
#define BS 40

template <int MODE>
__global__ __launch_bounds__(128) void gemm_wmma(
    const void* __restrict__ Ap, const float* __restrict__ Bw,
    const float* __restrict__ bias, float* __restrict__ outF,
    unsigned short* __restrict__ qb, unsigned short* __restrict__ kb,
    unsigned short* __restrict__ vb, int M, int N, int K) {
  __shared__ __align__(16) unsigned short At[64 * AS];
  __shared__ __align__(16) unsigned short Bt[64 * BS];  // [col][k] (transposed)

  const int t = threadIdx.x;
  const int wave = t >> 5, lane = t & 31, l15 = lane & 15, half = lane >> 4;
  const int wm = (wave >> 1) * 32, wn = (wave & 1) * 32;
  const int n0 = blockIdx.x * 64, m0 = blockIdx.y * 64;

  v8f acc[2][2];
#pragma unroll
  for (int i = 0; i < 2; ++i)
#pragma unroll
    for (int j = 0; j < 2; ++j)
#pragma unroll
      for (int r = 0; r < 8; ++r) acc[i][j][r] = 0.f;

  for (int k0 = 0; k0 < K; k0 += 32) {
    // ---- stage A tile (64 x 32) as bf16 ----
    if (MODE == 0) {
      const float* A = (const float*)Ap;
#pragma unroll
      for (int it = 0; it < 4; ++it) {
        int idx = t + it * 128;                 // 512 float4 total
        int row = idx >> 3, c4 = (idx & 7) * 4;
        float4 v = *(const float4*)(A + (size_t)(m0 + row) * K + k0 + c4);
        uint2 p;
        p.x = (unsigned int)f2bf(v.x) | ((unsigned int)f2bf(v.y) << 16);
        p.y = (unsigned int)f2bf(v.z) | ((unsigned int)f2bf(v.w) << 16);
        *(uint2*)&At[row * AS + c4] = p;
      }
    } else {
      const unsigned short* A = (const unsigned short*)Ap;
#pragma unroll
      for (int it = 0; it < 2; ++it) {
        int idx = t + it * 128;                 // 256 x 16B total
        int row = idx >> 2, c8 = (idx & 3) * 8;
        async_copy_b128(A + (size_t)(m0 + row) * K + k0 + c8,
                        &At[row * AS + c8]);
      }
    }
    // ---- stage B tile transposed: Bt[col][k] ----
#pragma unroll
    for (int it = 0; it < 4; ++it) {
      int idx = t + it * 128;                   // 512 float4 total
      int kr = idx >> 4, c4 = (idx & 15) * 4;
      float4 v = *(const float4*)(Bw + (size_t)(k0 + kr) * N + n0 + c4);
      Bt[(c4 + 0) * BS + kr] = f2bf(v.x);
      Bt[(c4 + 1) * BS + kr] = f2bf(v.y);
      Bt[(c4 + 2) * BS + kr] = f2bf(v.z);
      Bt[(c4 + 3) * BS + kr] = f2bf(v.w);
    }
    if (MODE == 1) wait_async0();  // drain async A-tile copies
    __syncthreads();

    // ---- build fragments, 4 WMMAs per wave per k-step ----
    v16bf af[2], bf[2];
#pragma unroll
    for (int mi = 0; mi < 2; ++mi) {
      int row = wm + mi * 16 + l15;   // A layout: K chunks at 8h and 16+8h
      af[mi] = frag_from(*(const uint4*)&At[row * AS + 8 * half],
                         *(const uint4*)&At[row * AS + 16 + 8 * half]);
    }
#pragma unroll
    for (int ni = 0; ni < 2; ++ni) {
      int col = wn + ni * 16 + l15;   // B layout: K chunks at 8h and 16+8h
      bf[ni] = frag_from(*(const uint4*)&Bt[col * BS + 8 * half],
                         *(const uint4*)&Bt[col * BS + 16 + 8 * half]);
    }
#pragma unroll
    for (int mi = 0; mi < 2; ++mi)
#pragma unroll
      for (int ni = 0; ni < 2; ++ni)
        acc[mi][ni] = __builtin_amdgcn_wmma_f32_16x16x32_bf16(
            false, af[mi], false, bf[ni], (short)0, acc[mi][ni], false, false);
    __syncthreads();
  }

  // ---- epilogue ----
#pragma unroll
  for (int mi = 0; mi < 2; ++mi) {
#pragma unroll
    for (int ni = 0; ni < 2; ++ni) {
      int c = n0 + wn + ni * 16 + l15;
      float bv = bias[c];
#pragma unroll
      for (int r = 0; r < 8; ++r) {
        int row = m0 + wm + mi * 16 + r + 8 * half;  // C layout: m = r + 8*half
        float v = acc[mi][ni][r] + bv;
        if (MODE == 0) {
          int which = c >> 10;        // 0:q 1:k 2:v
          int e = c & 1023;
          int h = e >> 6, d = e & 63;
          int b = row >> 11;          // row / SEQ
          int l = row & (SEQ - 1);
          unsigned short* dst = which == 0 ? qb : (which == 1 ? kb : vb);
          dst[(((size_t)(b * NHEADS + h)) * SEQ + l) * HDIM + d] = f2bf(v);
        } else {
          outF[(size_t)row * N + c] = v;
        }
      }
    }
  }
}

// ---------------------------------------------------------------------------
// Per-row squared norms of Q and K (bf16 in, f32 out).
// ---------------------------------------------------------------------------
__global__ __launch_bounds__(256) void row_norms(
    const unsigned short* __restrict__ qb, const unsigned short* __restrict__ kb,
    float* __restrict__ qn, float* __restrict__ kn, int T) {
  int idx = blockIdx.x * 256 + threadIdx.x;
  if (idx >= 2 * T) return;
  bool isq = idx < T;
  int r = isq ? idx : idx - T;
  const unsigned short* src = (isq ? qb : kb) + (size_t)r * HDIM;
  float s = 0.f;
#pragma unroll
  for (int i = 0; i < 8; ++i) {
    uint4 u = ((const uint4*)src)[i];
    unsigned int w[4] = {u.x, u.y, u.z, u.w};
#pragma unroll
    for (int j = 0; j < 4; ++j) {
      float a = bf2f((unsigned short)(w[j] & 0xFFFF));
      float b = bf2f((unsigned short)(w[j] >> 16));
      s += a * a + b * b;
    }
  }
  (isq ? qn : kn)[r] = s;
}

// ---------------------------------------------------------------------------
// Flash-style yat attention. Block = one (b,h) x 64 rows, 4 waves x 16 rows.
// K tile staged via async global->LDS DMA; V tile transposed through VGPRs.
// ---------------------------------------------------------------------------
#define KS 72   // padded LDS strides (shorts)
#define PS 72

__global__ __launch_bounds__(128) void yat_attention(
    const unsigned short* __restrict__ qb, const unsigned short* __restrict__ kb,
    const unsigned short* __restrict__ vb, const float* __restrict__ qnorm,
    const float* __restrict__ knorm, unsigned short* __restrict__ attn_out) {
  __shared__ __align__(16) unsigned short Kt[64 * KS];      // [col][d]
  __shared__ __align__(16) unsigned short Vt[64 * KS];      // [d][col] (transposed)
  __shared__ __align__(16) unsigned short Pb[4][16 * PS];   // per-wave P tile

  const int t = threadIdx.x;
  const int wave = t >> 5, lane = t & 31, l15 = lane & 15, half = lane >> 4;
  const int ntiles = SEQ / 64;
  const int tile = blockIdx.x % ntiles;
  const int bh = blockIdx.x / ntiles;
  const int b = bh / NHEADS, h = bh % NHEADS;
  const size_t headbase = (size_t)bh * SEQ * HDIM;
  const int rowbase = tile * 64 + wave * 16;

  // Q fragments (16 rows x D=64 -> two 16x32 A-fragments), direct from global.
  v16bf qf[2];
  {
    const unsigned short* qrow = qb + headbase + (size_t)(rowbase + l15) * HDIM;
#pragma unroll
    for (int ks = 0; ks < 2; ++ks)
      qf[ks] = frag_from(*(const uint4*)(qrow + 32 * ks + 8 * half),
                         *(const uint4*)(qrow + 32 * ks + 16 + 8 * half));
  }
  float qn[8];
#pragma unroll
  for (int r = 0; r < 8; ++r)
    qn[r] = qnorm[(size_t)bh * SEQ + rowbase + r + 8 * half];

  v8f acc[4];
#pragma unroll
  for (int i = 0; i < 4; ++i)
#pragma unroll
    for (int r = 0; r < 8; ++r) acc[i][r] = 0.f;
  float mrun[8], lrun[8];
#pragma unroll
  for (int r = 0; r < 8; ++r) { mrun[r] = -1e30f; lrun[r] = 0.f; }

  for (int j0 = 0; j0 <= tile * 64; j0 += 64) {
    // ---- stage K via async DMA, V (transposed) via VGPRs ----
#pragma unroll
    for (int it = 0; it < 4; ++it) {
      int idx = t + it * 128;                   // 512 x 16B = 64x64 bf16
      int row = idx >> 3, c8 = (idx & 7) * 8;
      size_t g = headbase + (size_t)(j0 + row) * HDIM + c8;
      async_copy_b128(kb + g, &Kt[row * KS + c8]);
      uint4 v = *(const uint4*)(vb + g);
      if (j0 + 64 <= tile * 64)                 // prefetch next tile
        __builtin_prefetch(kb + g + 64 * HDIM, 0, 1);
      unsigned int w[4] = {v.x, v.y, v.z, v.w};
#pragma unroll
      for (int j = 0; j < 4; ++j) {
        Vt[(c8 + 2 * j + 0) * KS + row] = (unsigned short)(w[j] & 0xFFFF);
        Vt[(c8 + 2 * j + 1) * KS + row] = (unsigned short)(w[j] >> 16);
      }
    }
    wait_async0();                              // K tile landed in LDS
    __syncthreads();

    // ---- S = Q * K^T : 16 rows x 64 cols (8 WMMAs) ----
    v8f sf[4];
#pragma unroll
    for (int nb = 0; nb < 4; ++nb) {
#pragma unroll
      for (int r = 0; r < 8; ++r) sf[nb][r] = 0.f;
      int col = nb * 16 + l15;
#pragma unroll
      for (int ks = 0; ks < 2; ++ks) {
        v16bf kf =
            frag_from(*(const uint4*)&Kt[col * KS + 32 * ks + 8 * half],
                      *(const uint4*)&Kt[col * KS + 32 * ks + 16 + 8 * half]);
        sf[nb] = __builtin_amdgcn_wmma_f32_16x16x32_bf16(
            false, qf[ks], false, kf, (short)0, sf[nb], false, false);
      }
    }

    // ---- yat score + causal mask (f32) ----
    float knc[4];
#pragma unroll
    for (int nb = 0; nb < 4; ++nb)
      knc[nb] = knorm[(size_t)bh * SEQ + j0 + nb * 16 + l15];
#pragma unroll
    for (int nb = 0; nb < 4; ++nb) {
      int gcol = j0 + nb * 16 + l15;
#pragma unroll
      for (int r = 0; r < 8; ++r) {
        int grow = rowbase + r + 8 * half;
        float dot = sf[nb][r];
        float denom = qn[r] + knc[nb] - 2.f * dot + EPS;
        float s = dot * dot / denom;
        sf[nb][r] = (gcol <= grow) ? s : -1e30f;
      }
    }

    // ---- online softmax (rows replicated across 16-lane halves) ----
    float alpha[8];
#pragma unroll
    for (int r = 0; r < 8; ++r) {
      float m = fmaxf(fmaxf(sf[0][r], sf[1][r]), fmaxf(sf[2][r], sf[3][r]));
#pragma unroll
      for (int d = 1; d < 16; d <<= 1) m = fmaxf(m, __shfl_xor(m, d, 32));
      float mnew = fmaxf(mrun[r], m);
      alpha[r] = __expf(mrun[r] - mnew);
      mrun[r] = mnew;
    }
#pragma unroll
    for (int nb = 0; nb < 4; ++nb)
#pragma unroll
      for (int r = 0; r < 8; ++r) sf[nb][r] = __expf(sf[nb][r] - mrun[r]);
#pragma unroll
    for (int r = 0; r < 8; ++r) {
      float s = sf[0][r] + sf[1][r] + sf[2][r] + sf[3][r];
#pragma unroll
      for (int d = 1; d < 16; d <<= 1) s += __shfl_xor(s, d, 32);
      lrun[r] = lrun[r] * alpha[r] + s;
    }
#pragma unroll
    for (int i = 0; i < 4; ++i)
#pragma unroll
      for (int r = 0; r < 8; ++r) acc[i][r] *= alpha[r];

    // ---- P (C layout) -> per-wave LDS -> A-fragment layout ----
    unsigned short* P = Pb[wave];
#pragma unroll
    for (int nb = 0; nb < 4; ++nb)
#pragma unroll
      for (int r = 0; r < 8; ++r)
        P[(r + 8 * half) * PS + nb * 16 + l15] = f2bf(sf[nb][r]);
    asm volatile("s_wait_dscnt 0" ::: "memory");  // DS is in-order per wave

    // ---- O += P * V (8 WMMAs) ----
#pragma unroll
    for (int ks = 0; ks < 2; ++ks) {
      v16bf pf =
          frag_from(*(const uint4*)&P[l15 * PS + 32 * ks + 8 * half],
                    *(const uint4*)&P[l15 * PS + 32 * ks + 16 + 8 * half]);
#pragma unroll
      for (int nbd = 0; nbd < 4; ++nbd) {
        int d = nbd * 16 + l15;
        v16bf vf =
            frag_from(*(const uint4*)&Vt[d * KS + 32 * ks + 8 * half],
                      *(const uint4*)&Vt[d * KS + 32 * ks + 16 + 8 * half]);
        acc[nbd] = __builtin_amdgcn_wmma_f32_16x16x32_bf16(
            false, pf, false, vf, (short)0, acc[nbd], false, false);
      }
    }
    __syncthreads();  // Kt/Vt reused next iteration
  }

  // ---- normalize and store attn output (bf16, [B*L][E] with col = h*64+d) ----
  float rl[8];
#pragma unroll
  for (int r = 0; r < 8; ++r) rl[r] = 1.0f / lrun[r];
#pragma unroll
  for (int nbd = 0; nbd < 4; ++nbd) {
    int d = nbd * 16 + l15;
#pragma unroll
    for (int r = 0; r < 8; ++r) {
      int row = rowbase + r + 8 * half;
      attn_out[((size_t)(b * SEQ + row)) * EMBED + h * HDIM + d] =
          f2bf(acc[nbd][r] * rl[r]);
    }
  }
}

// ---------------------------------------------------------------------------
extern "C" void kernel_launch(void* const* d_in, const int* in_sizes, int n_in,
                              void* d_out, int out_size, void* d_ws,
                              size_t ws_size, hipStream_t stream) {
  (void)in_sizes; (void)n_in; (void)out_size; (void)ws_size;
  const float* x     = (const float*)d_in[0];
  const float* w_qkv = (const float*)d_in[1];
  const float* b_qkv = (const float*)d_in[2];
  const float* w_out = (const float*)d_in[3];
  const float* b_out = (const float*)d_in[4];
  float* out = (float*)d_out;

  const size_t HBUF = (size_t)BATCH * NHEADS * SEQ * HDIM;  // 4,194,304 elems
  unsigned short* qb   = (unsigned short*)d_ws;
  unsigned short* kb   = qb + HBUF;
  unsigned short* vb   = kb + HBUF;
  unsigned short* attn = vb + HBUF;                          // B*L*E, same count
  float* qn = (float*)(attn + HBUF);
  float* kn = qn + (size_t)BATCH * NHEADS * SEQ;

  const int M = BATCH * SEQ;           // 4096
  const int T = BATCH * NHEADS * SEQ;  // 65536 rows each for Q/K norms

  // 1) QKV projection -> bf16 Q/K/V in [b,h,l,d]
  gemm_wmma<0><<<dim3(3 * EMBED / 64, M / 64), 128, 0, stream>>>(
      (const void*)x, w_qkv, b_qkv, nullptr, qb, kb, vb, M, 3 * EMBED, EMBED);

  // 2) row norms
  row_norms<<<(2 * T + 255) / 256, 256, 0, stream>>>(qb, kb, qn, kn, T);

  // 3) flash yat attention
  yat_attention<<<BATCH * NHEADS * (SEQ / 64), 128, 0, stream>>>(
      qb, kb, vb, qn, kn, attn);

  // 4) output projection -> fp32 d_out
  gemm_wmma<1><<<dim3(EMBED / 64, M / 64), 128, 0, stream>>>(
      (const void*)attn, w_out, b_out, out, nullptr, nullptr, nullptr, M, EMBED,
      EMBED);
}
